// CausalSelfAttention_300647710884
// MI455X (gfx1250) — matrix operand order repaired and checked by reading, not compiled, not run on hardware
//
#include <hip/hip_runtime.h>
#include <hip/hip_bf16.h>

typedef __attribute__((ext_vector_type(16))) __bf16 v16bf;
typedef __attribute__((ext_vector_type(8)))  float  v8f;
typedef unsigned short u16;
typedef unsigned int   u32;

// ---------------- helpers ----------------

__device__ __forceinline__ u16 f2bf(float f) {
    u32 u = __builtin_bit_cast(u32, f);
    u32 r = u + 0x7FFFu + ((u >> 16) & 1u);   // round-to-nearest-even
    return (u16)(r >> 16);
}

__device__ __forceinline__ __bf16 bits2bf(u16 b) {
    return __builtin_bit_cast(__bf16, b);
}

// LDS byte offset of a __shared__ object: low 32 bits of the flat address
// (see ISA 10.2, LDS aperture).
__device__ __forceinline__ u32 lds_off(const void* p) {
    return (u32)(unsigned long long)p;
}

// CDNA5 async global->LDS copy, 16B per lane (tracked by ASYNCcnt).
__device__ __forceinline__ void async_copy_b128(u32 lds_byte_off, const void* gaddr) {
    asm volatile("global_load_async_to_lds_b128 %0, %1, off"
                 :: "v"(lds_byte_off), "v"(gaddr) : "memory");
}

__device__ __forceinline__ void wait_async0() {
    asm volatile("s_wait_asynccnt 0x0" ::: "memory");
}

// A-matrix fragment (16x32 bf16), per ISA 7.12.2:
// lane L: row m = L%16; hi = L/16.
// VGPR v(0..3): k = kbase + 8*hi + 2v + o ; VGPR v(4..7): k = kbase + 16 + 8*hi + 2(v-4) + o
__device__ __forceinline__ v16bf load_a_frag(const u16* base, int ld, int m, int hi, int kbase) {
    const u16* row = base + m * ld + kbase + 8 * hi;
    v16bf f;
#pragma unroll
    for (int v = 0; v < 4; ++v) {
        u32 p = *(const u32*)(row + 2 * v);
        f[2 * v]     = bits2bf((u16)(p & 0xFFFFu));
        f[2 * v + 1] = bits2bf((u16)(p >> 16));
    }
#pragma unroll
    for (int v = 0; v < 4; ++v) {
        u32 p = *(const u32*)(row + 16 + 2 * v);
        f[8 + 2 * v] = bits2bf((u16)(p & 0xFFFFu));
        f[9 + 2 * v] = bits2bf((u16)(p >> 16));
    }
    return f;
}

// B-matrix fragment (32x16 bf16) gathered from a K-transposed tile laid out [n][k]:
// lane L: col n = L%16; hi = L/16; element j<8: k = kbase + 8*hi + j ; j>=8: k = kbase + 16 + 8*hi + (j-8)
__device__ __forceinline__ v16bf load_bt_frag(const u16* base, int ld, int n, int hi, int kbase) {
    const u16* row = base + n * ld + kbase + 8 * hi;
    v16bf f;
#pragma unroll
    for (int v = 0; v < 4; ++v) {
        u32 p = *(const u32*)(row + 2 * v);
        f[2 * v]     = bits2bf((u16)(p & 0xFFFFu));
        f[2 * v + 1] = bits2bf((u16)(p >> 16));
    }
#pragma unroll
    for (int v = 0; v < 4; ++v) {
        u32 p = *(const u32*)(row + 16 + 2 * v);
        f[8 + 2 * v] = bits2bf((u16)(p & 0xFFFFu));
        f[9 + 2 * v] = bits2bf((u16)(p >> 16));
    }
    return f;
}

__device__ __forceinline__ v8f wmma_bf16(v16bf a, v16bf b, v8f c) {
    return __builtin_amdgcn_wmma_f32_16x16x32_bf16(
        /*neg_a=*/false, a, /*neg_b=*/false, b,
        /*c_mod=*/(short)0, c, /*reuse_a=*/false, /*reuse_b=*/false);
}

// GEMM staging: A tile 64x32 (pure copy, async) ; B tile 32x64 -> Bt[n][k] (transpose in flight)
__device__ __forceinline__ void stage_a64x32(u32 lds_base, const u16* g, int ldg, int tid) {
#pragma unroll
    for (int it = 0; it < 2; ++it) {
        int idx = tid + it * 128;                 // 16B chunk id (256 total)
        int r = idx >> 2, c8 = (idx & 3) * 8;
        async_copy_b128(lds_base + (u32)(idx * 16), g + r * ldg + c8);
    }
}

__device__ __forceinline__ void stage_bt64x32(u16* bt, const u16* g, int ldg, int tid) {
#pragma unroll
    for (int it = 0; it < 8; ++it) {
        int e   = tid + it * 128;                 // uint-pair id (1024 total)
        int idx = e * 2;
        int kk = idx >> 6, n = idx & 63;
        u32 p = *(const u32*)(g + kk * ldg + n);
        bt[n * 32 + kk]       = (u16)(p & 0xFFFFu);
        bt[(n + 1) * 32 + kk] = (u16)(p >> 16);
    }
}

// ---------------- kernel 1: fp32 -> bf16 ----------------

__global__ __launch_bounds__(256) void k_cvt(const float* __restrict__ s,
                                             u16* __restrict__ d, int n) {
    int i = (blockIdx.x * blockDim.x + threadIdx.x) * 4;
    if (i < n) {
        u16 t0 = f2bf(s[i]);
        u16 t1 = f2bf(s[i + 1]);
        u16 t2 = f2bf(s[i + 2]);
        u16 t3 = f2bf(s[i + 3]);
        *(u32*)(d + i)     = (u32)t0 | ((u32)t1 << 16);
        *(u32*)(d + i + 2) = (u32)t2 | ((u32)t3 << 16);
    }
}

// ---------------- kernel 2: QKV projection GEMM (double-buffered async pipeline) ----------------
// C[4096 x 3072] = Xb[4096 x 1024] @ Wb[1024 x 3072] + bias ; scatter into Q/K/V [B*H][S][64] bf16

__global__ __launch_bounds__(128) void k_qkv_gemm(const u16* __restrict__ X,
                                                  const u16* __restrict__ W,
                                                  const float* __restrict__ bias,
                                                  u16* __restrict__ Q,
                                                  u16* __restrict__ K,
                                                  u16* __restrict__ V) {
    __shared__ __align__(16) u16 As[2][64 * 32];   // [m][k]
    __shared__ __align__(16) u16 Bt[2][64 * 32];   // [n][k]
    const int tid  = threadIdx.x;
    const int wid  = tid >> 5;
    const int lane = tid & 31;
    const int ln   = lane & 15;
    const int hi   = lane >> 4;
    const int mbase = blockIdx.y * 64;
    const int nbase = blockIdx.x * 64;
    const u32 as_base[2] = { lds_off(As[0]), lds_off(As[1]) };

    v8f acc[4] = {};

    // prologue: stage tile 0
    stage_a64x32(as_base[0], X + mbase * 1024, 1024, tid);
    stage_bt64x32(Bt[0], W + nbase, 3072, tid);

    for (int k0 = 0; k0 < 1024; k0 += 32) {
        const int cb = (k0 >> 5) & 1;
        wait_async0();           // tile-cb async fills done (had full prev compute to overlap)
        __syncthreads();         // all waves' fills visible; all prior reads of buffer cb^1 done

        if (k0 + 32 < 1024) {    // stage next tile into the other buffer, overlapped with compute
            stage_a64x32(as_base[cb ^ 1], X + mbase * 1024 + (k0 + 32), 1024, tid);
            stage_bt64x32(Bt[cb ^ 1], W + (k0 + 32) * 3072 + nbase, 3072, tid);
            if (k0 + 64 < 1024)
                __builtin_prefetch(W + (k0 + 64) * 3072 + nbase, 0, 0);
        }

        v16bf a = load_a_frag(As[cb], 32, wid * 16 + ln, hi, 0);
#pragma unroll
        for (int ns = 0; ns < 4; ++ns) {
            v16bf b = load_bt_frag(Bt[cb], 32, ns * 16 + ln, hi, 0);
            acc[ns] = wmma_bf16(a, b, acc[ns]);
        }
    }

#pragma unroll
    for (int ns = 0; ns < 4; ++ns) {
        int col   = nbase + ns * 16 + ln;
        int three = col >> 10;
        int rem   = col & 1023;
        int h     = rem >> 6;
        int d     = rem & 63;
        float bs  = bias[col];
        u16* dst  = (three == 0) ? Q : (three == 1) ? K : V;
        float sc  = (three == 0) ? 0.125f : 1.0f;      // fold 1/sqrt(HD) into Q
#pragma unroll
        for (int r = 0; r < 8; ++r) {
            int row = mbase + wid * 16 + r + 8 * hi;   // global row in [0,4096)
            int b_  = row >> 11;
            int s_  = row & 2047;
            float val = (acc[ns][r] + bs) * sc;
            dst[((b_ * 16 + h) * 2048 + s_) * 64 + d] = f2bf(val);
        }
    }
}

// ---------------- kernel 3: fused flash attention (double-buffered K/V tiles) ----------------
// grid: (S/64, B*H); block: 128 (4 waves); each wave owns 16 q rows.

__device__ __forceinline__ void stage_ktile(u32 lds_base, const u16* kg, int tid) {
#pragma unroll
    for (int it = 0; it < 4; ++it) {               // 8 KiB contiguous copy
        int idx = tid + it * 128;
        async_copy_b128(lds_base + (u32)(idx * 16), kg + idx * 8);
    }
}

__device__ __forceinline__ void stage_vtile(u16* vt, const u16* vg, int tid) {
#pragma unroll
    for (int it = 0; it < 16; ++it) {              // transpose into [d][key]
        int e   = tid + it * 128;
        int idx = e * 2;
        int key = idx >> 6, d2 = idx & 63;
        u32 p = *(const u32*)(vg + key * 64 + d2);
        vt[d2 * 64 + key]       = (u16)(p & 0xFFFFu);
        vt[(d2 + 1) * 64 + key] = (u16)(p >> 16);
    }
}

__global__ __launch_bounds__(128) void k_attn(const u16* __restrict__ Q,
                                              const u16* __restrict__ Kb,
                                              const u16* __restrict__ Vb,
                                              u16* __restrict__ Ob) {
    __shared__ __align__(16) u16 Kt[2][64 * 64];    // [key][d]  (== transposed-B for Q@K^T)
    __shared__ __align__(16) u16 Vt[2][64 * 64];    // [d][key]  (transposed-B for P@V)
    __shared__ __align__(16) u16 Ps[4][16 * 64];    // per-wave P staging (row-major 16x64)

    const int tid  = threadIdx.x;
    const int wid  = tid >> 5;
    const int lane = tid & 31;
    const int ln   = lane & 15;
    const int hi   = lane >> 4;
    const int qt    = blockIdx.x;
    const int bh    = blockIdx.y;
    const int qbase = qt * 64;
    const u32 kt_base[2] = { lds_off(Kt[0]), lds_off(Kt[1]) };

    // Q fragments held in registers for the whole kernel (16 rows x 64 dim)
    const u16* qptr = Q + (bh * 2048 + qbase + wid * 16) * 64;
    v16bf qa0 = load_a_frag(qptr, 64, ln, hi, 0);
    v16bf qa1 = load_a_frag(qptr, 64, ln, hi, 32);

    v8f o[4] = {};
    float l[8], mx[8];
#pragma unroll
    for (int r = 0; r < 8; ++r) { l[r] = 0.0f; mx[r] = -3.0e38f; }

    const u16* kbh = Kb + bh * 2048 * 64;
    const u16* vbh = Vb + bh * 2048 * 64;
    const int ntiles = qt + 1;

    // prologue: stage key-tile 0
    stage_ktile(kt_base[0], kbh, tid);
    stage_vtile(Vt[0], vbh, tid);

    for (int t = 0; t < ntiles; ++t) {
        const int cb = t & 1;
        const int kt = t * 64;
        wait_async0();
        __syncthreads();

        if (t + 1 < ntiles) {   // stage next key/value tile while computing this one
            stage_ktile(kt_base[cb ^ 1], kbh + (kt + 64) * 64, tid);
            stage_vtile(Vt[cb ^ 1], vbh + (kt + 64) * 64, tid);
        }

        // scores S = Q @ K^T  (already scaled via Q)
        v8f sa[4];
#pragma unroll
        for (int ns = 0; ns < 4; ++ns) {
            v8f z = {};
            v16bf b0 = load_bt_frag(Kt[cb], 64, ns * 16 + ln, hi, 0);
            z = wmma_bf16(qa0, b0, z);
            v16bf b1 = load_bt_frag(Kt[cb], 64, ns * 16 + ln, hi, 32);
            sa[ns] = wmma_bf16(qa1, b1, z);
        }

        // causal mask + online softmax (row r+8*hi lives across 16 lanes of a half-wave)
#pragma unroll
        for (int r = 0; r < 8; ++r) {
            int qrow = qbase + wid * 16 + r + 8 * hi;
            float scm[4];
            float rowmax = -3.0e38f;
#pragma unroll
            for (int ns = 0; ns < 4; ++ns) {
                int key = kt + ns * 16 + ln;
                float sc = sa[ns][r];
                if (key > qrow) sc = -1.0e9f;
                scm[ns] = sc;
                rowmax = fmaxf(rowmax, sc);
            }
#pragma unroll
            for (int off = 8; off; off >>= 1)
                rowmax = fmaxf(rowmax, __shfl_xor(rowmax, off));

            float mnew = fmaxf(mx[r], rowmax);
            float corr = __expf(mx[r] - mnew);
            mx[r] = mnew;

            float rs = 0.0f;
#pragma unroll
            for (int ns = 0; ns < 4; ++ns) {
                float p = __expf(scm[ns] - mnew);
                rs += p;
                Ps[wid][(r + 8 * hi) * 64 + ns * 16 + ln] = f2bf(p);
            }
#pragma unroll
            for (int off = 8; off; off >>= 1)
                rs += __shfl_xor(rs, off);
            l[r] = l[r] * corr + rs;
#pragma unroll
            for (int ds = 0; ds < 4; ++ds)
                o[ds][r] *= corr;
        }
        __syncthreads();   // P staging visible (uniform control flow across waves)

        // O += P @ V
        v16bf p0 = load_a_frag(&Ps[wid][0], 64, ln, hi, 0);
        v16bf p1 = load_a_frag(&Ps[wid][0], 64, ln, hi, 32);
#pragma unroll
        for (int ds = 0; ds < 4; ++ds) {
            v16bf b0 = load_bt_frag(Vt[cb], 64, ds * 16 + ln, hi, 0);
            o[ds] = wmma_bf16(p0, b0, o[ds]);
            v16bf b1 = load_bt_frag(Vt[cb], 64, ds * 16 + ln, hi, 32);
            o[ds] = wmma_bf16(p1, b1, o[ds]);
        }
    }

    // normalize and scatter to attn_out [B*S][1024] (col = h*64 + d), bf16
    int b_ = bh >> 4, h = bh & 15;
#pragma unroll
    for (int ds = 0; ds < 4; ++ds) {
#pragma unroll
        for (int r = 0; r < 8; ++r) {
            int qrow = qbase + wid * 16 + r + 8 * hi;
            float val = o[ds][r] / l[r];
            Ob[(b_ * 2048 + qrow) * 1024 + h * 64 + ds * 16 + ln] = f2bf(val);
        }
    }
}

// ---------------- kernel 4: output projection (double-buffered async pipeline) ----------------
// out[4096 x 1024] = Ab[4096 x 1024] @ Wb[1024 x 1024] + bout  (fp32 result)

__global__ __launch_bounds__(128) void k_out_gemm(const u16* __restrict__ A,
                                                  const u16* __restrict__ W,
                                                  const float* __restrict__ bias,
                                                  float* __restrict__ out) {
    __shared__ __align__(16) u16 As[2][64 * 32];
    __shared__ __align__(16) u16 Bt[2][64 * 32];
    const int tid  = threadIdx.x;
    const int wid  = tid >> 5;
    const int lane = tid & 31;
    const int ln   = lane & 15;
    const int hi   = lane >> 4;
    const int mbase = blockIdx.y * 64;
    const int nbase = blockIdx.x * 64;
    const u32 as_base[2] = { lds_off(As[0]), lds_off(As[1]) };

    v8f acc[4] = {};

    stage_a64x32(as_base[0], A + mbase * 1024, 1024, tid);
    stage_bt64x32(Bt[0], W + nbase, 1024, tid);

    for (int k0 = 0; k0 < 1024; k0 += 32) {
        const int cb = (k0 >> 5) & 1;
        wait_async0();
        __syncthreads();

        if (k0 + 32 < 1024) {
            stage_a64x32(as_base[cb ^ 1], A + mbase * 1024 + (k0 + 32), 1024, tid);
            stage_bt64x32(Bt[cb ^ 1], W + (k0 + 32) * 1024 + nbase, 1024, tid);
            if (k0 + 64 < 1024)
                __builtin_prefetch(W + (k0 + 64) * 1024 + nbase, 0, 0);
        }

        v16bf a = load_a_frag(As[cb], 32, wid * 16 + ln, hi, 0);
#pragma unroll
        for (int ns = 0; ns < 4; ++ns) {
            v16bf b = load_bt_frag(Bt[cb], 32, ns * 16 + ln, hi, 0);
            acc[ns] = wmma_bf16(a, b, acc[ns]);
        }
    }

#pragma unroll
    for (int ns = 0; ns < 4; ++ns) {
        int col  = nbase + ns * 16 + ln;
        float bs = bias[col];
#pragma unroll
        for (int r = 0; r < 8; ++r) {
            int row = mbase + wid * 16 + r + 8 * hi;
            out[row * 1024 + col] = acc[ns][r] + bs;
        }
    }
}

// ---------------- launch ----------------

extern "C" void kernel_launch(void* const* d_in, const int* in_sizes, int n_in,
                              void* d_out, int out_size, void* d_ws, size_t ws_size,
                              hipStream_t stream) {
    (void)in_sizes; (void)n_in; (void)out_size; (void)ws_size;
    const float* x    = (const float*)d_in[0];
    // d_in[1] = mask (causal; applied analytically)
    const float* Wqkv = (const float*)d_in[2];
    const float* bqkv = (const float*)d_in[3];
    const float* Wout = (const float*)d_in[4];
    const float* bout = (const float*)d_in[5];
    float* out = (float*)d_out;

    char* ws = (char*)d_ws;
    u16* xb    = (u16*)(ws);                           // 8 MiB : x bf16 [4096][1024]
    u16* wqkvb = (u16*)(ws + (size_t)(8u  << 20));     // 6 MiB : Wqkv bf16 [1024][3072]
    u16* woutb = (u16*)(ws + (size_t)(14u << 20));     // 2 MiB : Wout bf16 [1024][1024]
    u16* qb    = (u16*)(ws + (size_t)(16u << 20));     // 8 MiB : Q bf16 [32][2048][64] (pre-scaled)
    u16* kb    = (u16*)(ws + (size_t)(24u << 20));     // 8 MiB : K bf16
    u16* vb    = (u16*)(ws + (size_t)(32u << 20));     // 8 MiB : V bf16
    u16* ab    = (u16*)(ws + (size_t)(40u << 20));     // 8 MiB : attn out bf16 [4096][1024]

    k_cvt<<<4096, 256, 0, stream>>>(x,    xb,    2 * 2048 * 1024);
    k_cvt<<<3072, 256, 0, stream>>>(Wqkv, wqkvb, 1024 * 3072);
    k_cvt<<<1024, 256, 0, stream>>>(Wout, woutb, 1024 * 1024);

    k_qkv_gemm<<<dim3(48, 64), 128, 0, stream>>>(xb, wqkvb, bqkv, qb, kb, vb);
    k_attn    <<<dim3(32, 32), 128, 0, stream>>>(qb, kb, vb, ab);
    k_out_gemm<<<dim3(16, 64), 128, 0, stream>>>(ab, woutb, bout, out);
}
